// MambaBlock_48850958025090
// MI455X (gfx1250) — matrix operand order repaired and checked by reading, not compiled, not run on hardware
//
#include <hip/hip_runtime.h>
#include <hip/hip_bf16.h>
#include <math.h>

// ---------------- problem constants ----------------
#define BATCH   8
#define SEQ     1024
#define DMODEL  256
#define DINNER  1024
#define DSTATE  32
#define DCONV   8
#define DTRANK  16
#define MROWS   (BATCH * SEQ)          // 8192
#define LN_EPS  1e-5f

// ---------------- types ----------------
typedef __attribute__((ext_vector_type(16))) __bf16 v16bf;
typedef __attribute__((ext_vector_type(8)))  float  v8f;

union FragU { int i[8]; v16bf vec; };

__device__ __forceinline__ unsigned short f2bf(float f) {
  unsigned int u = __float_as_uint(f);
  unsigned int r = u + 0x7FFFu + ((u >> 16) & 1u);   // round-to-nearest-even
  return (unsigned short)(r >> 16);
}

// A fragment (16-bit A 16x32): lanes0-15 K={0..7,16..23}, lanes16-31 K={8..15,24..31}.
// base = uniform pointer, off = per-lane element offset (row*lda + kh*8 + kb).
// Per-r displacements are dword immediates {0,1,2,3,8,9,10,11}.
__device__ __forceinline__ void load_fragA(FragU& f, const unsigned short* base, unsigned off) {
  const int* p = (const int*)(base + off);
  f.i[0] = p[0]; f.i[1] = p[1]; f.i[2] = p[2];  f.i[3] = p[3];
  f.i[4] = p[8]; f.i[5] = p[9]; f.i[6] = p[10]; f.i[7] = p[11];
}
// B fragment (B 32x16): lanes0-15 K={0..7,16..23}, lanes16-31 K={8..15,24..31}.
// off = col*ldb + kh*8 + kb; dwords {0,1,2,3,8,9,10,11}.
__device__ __forceinline__ void load_fragB(FragU& f, const unsigned short* base, unsigned off) {
  const int* p = (const int*)(base + off);
  f.i[0] = p[0]; f.i[1] = p[1]; f.i[2] = p[2]; f.i[3] = p[3];
  f.i[4] = p[8]; f.i[5] = p[9]; f.i[6] = p[10]; f.i[7] = p[11];
}

__device__ __forceinline__ void store_tile(const v8f acc, float* __restrict__ C, int ldc,
                                           const float* __restrict__ resid,
                                           int mo, int nn, int Nt) {
  if (nn < Nt) {
    #pragma unroll
    for (int r = 0; r < 8; ++r) {
      size_t idx = (size_t)(mo + r) * ldc + nn;
      float val = acc[r];
      if (resid) val += resid[idx];
      C[idx] = val;
    }
  }
}

// ---------------- fp32 -> bf16 conversion (with zero row padding) ----------------
__global__ void cvt_f32_bf16_kernel(const float* __restrict__ src,
                                    unsigned short* __restrict__ dst,
                                    int n_src, int n_total) {
  int i = blockIdx.x * 256 + threadIdx.x;
  if (i < n_total) dst[i] = (i < n_src) ? f2bf(src[i]) : (unsigned short)0;
}

// ---------------- LayerNorm: one wave (32 lanes) per 256-wide row ----------------
__global__ void layernorm_kernel(const float* __restrict__ x,
                                 const float* __restrict__ w,
                                 const float* __restrict__ b,
                                 unsigned short* __restrict__ xn_bf) {
  int row  = blockIdx.x * blockDim.y + threadIdx.y;   // 0..8191
  int lane = threadIdx.x;
  const float* xr = x + (size_t)row * DMODEL;
  float v[8];
  float s = 0.f;
  #pragma unroll
  for (int i = 0; i < 8; ++i) { v[i] = xr[i * 32 + lane]; s += v[i]; }
  #pragma unroll
  for (int off = 16; off >= 1; off >>= 1) s += __shfl_xor(s, off);
  float mu = s * (1.f / DMODEL);
  float vs = 0.f;
  #pragma unroll
  for (int i = 0; i < 8; ++i) { float d = v[i] - mu; vs += d * d; }
  #pragma unroll
  for (int off = 16; off >= 1; off >>= 1) vs += __shfl_xor(vs, off);
  float rstd = rsqrtf(vs * (1.f / DMODEL) + LN_EPS);
  unsigned short* o = xn_bf + (size_t)row * DMODEL;
  #pragma unroll
  for (int i = 0; i < 8; ++i) {
    int c = i * 32 + lane;
    o[c] = f2bf((v[i] - mu) * rstd * w[c] + b[c]);
  }
}

// ---------------- bf16 WMMA GEMM, 16x64 tile per wave, K double-buffered ----------------
// C[Mt,Nt] (row-major fp32, ld=ldc) = A[Mt,K] (row-major bf16, ld=lda)
//                                   x Wt[>=ceil64(Nt),K] (row-major bf16, ld=ldb) == A*W^T
// Wt zero-padded to a multiple of 64 rows. Stores predicated on nn<Nt.
// K must be a multiple of 64. All fragment loads use scalar-base + 32-bit lane
// offsets (GVS addressing) to keep VGPR pressure low (no 64-bit per-lane addrs).
__global__ void gemm_bf16_wmma(const unsigned short* __restrict__ A, int lda,
                               const unsigned short* __restrict__ Wt, int ldb,
                               float* __restrict__ C, int ldc,
                               const float* __restrict__ resid,
                               int Mt, int Nt, int K) {
  int ngroups = (Nt + 63) >> 6;                    // 64-wide N groups
  int total   = (Mt >> 4) * ngroups;
  int grp     = blockIdx.x * blockDim.y + threadIdx.y;  // uniform per wave
  if (grp >= total) return;
  int mt = (grp / ngroups) << 4;
  int n0 = (grp % ngroups) << 6;
  int lane = threadIdx.x;
  int l15  = lane & 15;
  int kh   = lane >> 4;

  // 32-bit per-lane element offsets (buffers < 2^31 elements)
  unsigned offA  = (unsigned)(mt + l15) * (unsigned)lda + (unsigned)(kh << 3);
  unsigned offB0 = (unsigned)(n0 + l15) * (unsigned)ldb + (unsigned)(kh << 3);
  unsigned sB    = (unsigned)ldb << 4;             // 16 rows of B
  unsigned offB1 = offB0 + sB;
  unsigned offB2 = offB1 + sB;
  unsigned offB3 = offB2 + sB;

  v8f acc0 = {}, acc1 = {}, acc2 = {}, acc3 = {};
  FragU faA, fb0A, fb1A, fb2A, fb3A;     // ping
  FragU faB, fb0B, fb1B, fb2B, fb3B;     // pong

  // prologue: chunk 0
  load_fragA(faA, A, offA);
  load_fragB(fb0A, Wt, offB0); load_fragB(fb1A, Wt, offB1);
  load_fragB(fb2A, Wt, offB2); load_fragB(fb3A, Wt, offB3);

  for (int kb = 0; kb + 64 <= K; kb += 64) {
    // loads for kb+32 in flight while WMMAs consume kb
    load_fragA(faB, A, offA + kb + 32);
    load_fragB(fb0B, Wt, offB0 + kb + 32); load_fragB(fb1B, Wt, offB1 + kb + 32);
    load_fragB(fb2B, Wt, offB2 + kb + 32); load_fragB(fb3B, Wt, offB3 + kb + 32);

    acc0 = __builtin_amdgcn_wmma_f32_16x16x32_bf16(false, faA.vec, false, fb0A.vec, (short)0, acc0, false, false);
    acc1 = __builtin_amdgcn_wmma_f32_16x16x32_bf16(false, faA.vec, false, fb1A.vec, (short)0, acc1, false, false);
    acc2 = __builtin_amdgcn_wmma_f32_16x16x32_bf16(false, faA.vec, false, fb2A.vec, (short)0, acc2, false, false);
    acc3 = __builtin_amdgcn_wmma_f32_16x16x32_bf16(false, faA.vec, false, fb3A.vec, (short)0, acc3, false, false);

    if (kb + 64 < K) {   // uniform guard: no OOB prefetch past K
      load_fragA(faA, A, offA + kb + 64);
      load_fragB(fb0A, Wt, offB0 + kb + 64); load_fragB(fb1A, Wt, offB1 + kb + 64);
      load_fragB(fb2A, Wt, offB2 + kb + 64); load_fragB(fb3A, Wt, offB3 + kb + 64);
    }

    acc0 = __builtin_amdgcn_wmma_f32_16x16x32_bf16(false, faB.vec, false, fb0B.vec, (short)0, acc0, false, false);
    acc1 = __builtin_amdgcn_wmma_f32_16x16x32_bf16(false, faB.vec, false, fb1B.vec, (short)0, acc1, false, false);
    acc2 = __builtin_amdgcn_wmma_f32_16x16x32_bf16(false, faB.vec, false, fb2B.vec, (short)0, acc2, false, false);
    acc3 = __builtin_amdgcn_wmma_f32_16x16x32_bf16(false, faB.vec, false, fb3B.vec, (short)0, acc3, false, false);
  }

  // C/D layout: lane -> n = lane&15; VGPR r -> m = r + 8*(lane>>4)
  int mo = mt + (kh << 3);
  store_tile(acc0, C, ldc, resid, mo, n0 + l15,      Nt);
  store_tile(acc1, C, ldc, resid, mo, n0 + 16 + l15, Nt);
  store_tile(acc2, C, ldc, resid, mo, n0 + 32 + l15, Nt);
  store_tile(acc3, C, ldc, resid, mo, n0 + 48 + l15, Nt);
}

// ---------------- causal depthwise conv (k=8) + bias + SiLU ----------------
__global__ void conv_silu_kernel(const float* __restrict__ xz,
                                 const float* __restrict__ convw,
                                 const float* __restrict__ convb,
                                 float* __restrict__ u32,
                                 unsigned short* __restrict__ ubf) {
  __shared__ float wsm[256 * DCONV];          // 8KB slice of conv weights
  int cblk = blockIdx.x & 3;                  // 4 c-blocks of 256
  int m    = blockIdx.x >> 2;                 // 0..8191
  int tid  = threadIdx.x;
  for (int i = tid; i < 256 * DCONV; i += 256)
    wsm[i] = convw[cblk * 256 * DCONV + i];
  __syncthreads();
  int c = cblk * 256 + tid;
  int b = m >> 10, l = m & 1023;
  float acc = convb[c];
  #pragma unroll
  for (int k = 0; k < DCONV; ++k) {
    int ls = l - (DCONV - 1) + k;
    if (ls >= 0)
      acc += xz[((size_t)((b << 10) + ls)) * (2 * DINNER) + c] * wsm[tid * DCONV + k];
  }
  float sil = acc * (1.f / (1.f + __expf(-acc)));
  size_t o = (size_t)m * DINNER + c;
  u32[o] = sil;
  ubf[o] = f2bf(sil);
}

// ---------------- dt_proj (K=16) + softplus ----------------
__global__ void dtproj_softplus_kernel(const float* __restrict__ xdbl,
                                       const float* __restrict__ dtw,
                                       const float* __restrict__ dtb,
                                       float* __restrict__ delta) {
  int d = blockIdx.y;                               // 0..1023 (uniform -> scalar weights)
  int m = blockIdx.x * blockDim.x + threadIdx.x;    // 0..8191
  float acc = dtb[d];
  const float* xr = xdbl + (size_t)m * 80;          // dt part = first 16 floats
  const float* wr = dtw + d * DTRANK;
  #pragma unroll
  for (int r = 0; r < DTRANK; ++r) acc += xr[r] * wr[r];
  float sp = (acc > 20.f) ? acc : log1pf(__expf(acc));
  delta[(size_t)m * DINNER + d] = sp;
}

// ---------------- selective scan ----------------
// one wave per (b,d); lane = state index n (wave32 == DSTATE). h lives in a register.
// fused: y = (h.C + u*D) * silu(z) -> bf16. Prefetch 16 steps ahead to shorten the
// serial critical path (global_prefetch_b8).
__global__ void selective_scan_kernel(const float* __restrict__ delta,
                                      const float* __restrict__ u32,
                                      const float* __restrict__ xdbl,
                                      const float* __restrict__ xz,
                                      const float* __restrict__ A_log,
                                      const float* __restrict__ Dvec,
                                      unsigned short* __restrict__ ybf) {
  int gw = blockIdx.x * blockDim.y + threadIdx.y;   // 0..8191
  int b = gw >> 10;
  int d = gw & 1023;
  int n = threadIdx.x;                              // lane = state index
  float An = -__expf(A_log[d * DSTATE + n]);
  float Dd = Dvec[d];
  float h = 0.f;
  size_t rowbase = (size_t)b * SEQ;
  for (int t = 0; t < SEQ; ++t) {
    size_t m = rowbase + t;
    if (t + 16 < SEQ) {                             // uniform branch
      size_t mp = m + 16;
      __builtin_prefetch(delta + mp * DINNER + d, 0, 3);
      __builtin_prefetch(u32 + mp * DINNER + d, 0, 3);
      __builtin_prefetch(xdbl + mp * 80 + DTRANK + n, 0, 3);
      __builtin_prefetch(xz + mp * (2 * DINNER) + DINNER + d, 0, 3);
    }
    float dt_ = delta[m * DINNER + d];
    float ut  = u32[m * DINNER + d];
    const float* xr = xdbl + m * 80;
    float Bn = xr[DTRANK + n];
    float Cn = xr[DTRANK + DSTATE + n];
    float dA = __expf(dt_ * An);
    h = dA * h + (dt_ * ut) * Bn;
    float yv = h * Cn;
    #pragma unroll
    for (int off = 16; off >= 1; off >>= 1) yv += __shfl_xor(yv, off);
    if (n == 0) {
      float z = xz[m * (2 * DINNER) + DINNER + d];
      float y = yv + ut * Dd;
      float g = z * (1.f / (1.f + __expf(-z)));
      ybf[m * DINNER + d] = f2bf(y * g);
    }
  }
}

// ---------------- host launch ----------------
extern "C" void kernel_launch(void* const* d_in, const int* in_sizes, int n_in,
                              void* d_out, int out_size, void* d_ws, size_t ws_size,
                              hipStream_t stream) {
  (void)in_sizes; (void)n_in; (void)out_size; (void)ws_size;
  const float* x        = (const float*)d_in[0];
  const float* norm_w   = (const float*)d_in[1];
  const float* norm_b   = (const float*)d_in[2];
  const float* in_projw = (const float*)d_in[3];
  const float* conv_w   = (const float*)d_in[4];
  const float* conv_b   = (const float*)d_in[5];
  const float* x_projw  = (const float*)d_in[6];
  const float* dt_projw = (const float*)d_in[7];
  const float* dt_projb = (const float*)d_in[8];
  const float* A_log    = (const float*)d_in[9];
  const float* Dvec     = (const float*)d_in[10];
  const float* out_projw= (const float*)d_in[11];
  float* out = (float*)d_out;

  // workspace layout
  char* ws = (char*)d_ws;
  size_t o = 0;
  unsigned short* xn_bf = (unsigned short*)(ws + o); o += (size_t)MROWS * DMODEL * 2;      // 4MB
  unsigned short* w_in  = (unsigned short*)(ws + o); o += (size_t)2 * DINNER * DMODEL * 2; // 1MB
  unsigned short* w_xp  = (unsigned short*)(ws + o); o += (size_t)128 * DINNER * 2;        // padded 80->128
  unsigned short* w_out = (unsigned short*)(ws + o); o += (size_t)DMODEL * DINNER * 2;
  float*          xz    = (float*)(ws + o);          o += (size_t)MROWS * 2 * DINNER * 4;  // 64MB
  float*          u32   = (float*)(ws + o);          o += (size_t)MROWS * DINNER * 4;      // 32MB
  unsigned short* u_bf  = (unsigned short*)(ws + o); o += (size_t)MROWS * DINNER * 2;      // 16MB
  float*          xdbl  = (float*)(ws + o);          o += (size_t)MROWS * 80 * 4;
  float*          delta = (float*)(ws + o);          o += (size_t)MROWS * DINNER * 4;      // 32MB
  unsigned short* y_bf  = (unsigned short*)(ws + o); o += (size_t)MROWS * DINNER * 2;      // 16MB

  // 1) weight conversions to bf16 (x_proj padded with zero rows to 128)
  {
    int n1 = 2 * DINNER * DMODEL;
    cvt_f32_bf16_kernel<<<(n1 + 255) / 256, 256, 0, stream>>>(in_projw, w_in, n1, n1);
    int n2s = 80 * DINNER, n2t = 128 * DINNER;
    cvt_f32_bf16_kernel<<<(n2t + 255) / 256, 256, 0, stream>>>(x_projw, w_xp, n2s, n2t);
    int n3 = DMODEL * DINNER;
    cvt_f32_bf16_kernel<<<(n3 + 255) / 256, 256, 0, stream>>>(out_projw, w_out, n3, n3);
  }

  // 2) LayerNorm -> bf16  (8 waves/block, one row per wave)
  layernorm_kernel<<<MROWS / 8, dim3(32, 8), 0, stream>>>(x, norm_w, norm_b, xn_bf);

  // 3) in_proj: xz[8192,2048] = xn * Win^T   (K=256)
  {
    int grps = (MROWS / 16) * ((2 * DINNER) / 64);
    gemm_bf16_wmma<<<(grps + 7) / 8, dim3(32, 8), 0, stream>>>(
        xn_bf, DMODEL, w_in, DMODEL, xz, 2 * DINNER, nullptr,
        MROWS, 2 * DINNER, DMODEL);
  }

  // 4) causal depthwise conv + bias + SiLU
  conv_silu_kernel<<<MROWS * 4, 256, 0, stream>>>(xz, conv_w, conv_b, u32, u_bf);

  // 5) x_proj: xdbl[8192,80] = u * Wxp^T   (K=1024); N padded to 128, stores guarded
  {
    int grps = (MROWS / 16) * (128 / 64);
    gemm_bf16_wmma<<<(grps + 7) / 8, dim3(32, 8), 0, stream>>>(
        u_bf, DINNER, w_xp, DINNER, xdbl, 80, nullptr,
        MROWS, 80, DINNER);
  }

  // 6) dt_proj (K=16) + softplus -> delta[8192,1024]
  dtproj_softplus_kernel<<<dim3(MROWS / 256, DINNER), 256, 0, stream>>>(
      xdbl, dt_projw, dt_projb, delta);

  // 7) selective scan + gating -> y bf16
  selective_scan_kernel<<<MROWS / 8, dim3(32, 8), 0, stream>>>(
      delta, u32, xdbl, xz, A_log, Dvec, y_bf);

  // 8) out_proj + residual: out[8192,256] = y * Wout^T + x   (K=1024)
  {
    int grps = (MROWS / 16) * (DMODEL / 64);
    gemm_bf16_wmma<<<(grps + 7) / 8, dim3(32, 8), 0, stream>>>(
        y_bf, DINNER, w_out, DINNER, out, DMODEL, x,
        MROWS, DMODEL, DINNER);
  }
}
